// MultiHeadAttentionMessagePassingNetwork_73589969649755
// MI455X (gfx1250) — matrix operand order repaired and checked by reading, not compiled, not run on hardware
//
#include <hip/hip_runtime.h>
#include <hip/hip_bf16.h>

// ---------------------------------------------------------------------------
// Types for CDNA5 WMMA (wave32, 16x16x32 bf16 -> f32)
// ---------------------------------------------------------------------------
typedef __attribute__((ext_vector_type(16))) __bf16 v16bf;
typedef __attribute__((ext_vector_type(8)))  float  v8f;

union BFrag {
    uint4 u[2];
    v16bf v;
};

#define HCH 128          // hidden channels
#define SA  136          // LDS row stride in bf16 elems (272B = 17*16B, aligned)
#define SLOPE 0.01f
#define LN_EPS 1e-5f
#define INV_TEMP 0.08838834764831845f   // 1/sqrt(128)

// ---- Tensor Data Mover availability (toolchain-guarded) -------------------
#if defined(__HIP_DEVICE_COMPILE__) && defined(__gfx1250__) && \
    __has_builtin(__builtin_amdgcn_tensor_load_to_lds) &&      \
    __has_builtin(__builtin_amdgcn_s_wait_tensorcnt)
#define USE_TDM 1
#else
#define USE_TDM 0
#endif

__device__ __forceinline__ unsigned short f2bf(float f) {
    unsigned u = __float_as_uint(f);
    unsigned r = u + 0x7fffu + ((u >> 16) & 1u);   // round to nearest even
    return (unsigned short)(r >> 16);
}
__device__ __forceinline__ float bf2f(unsigned short h) {
    return __uint_as_float(((unsigned)h) << 16);
}
__device__ __forceinline__ float leaky(float v) {
    return v >= 0.f ? v : SLOPE * v;
}
__device__ __forceinline__ unsigned fflip(float f) {     // order-preserving float->uint
    unsigned u = __float_as_uint(f);
    return (u & 0x80000000u) ? ~u : (u | 0x80000000u);
}
__device__ __forceinline__ float funflip(unsigned u) {
    return __uint_as_float((u & 0x80000000u) ? (u ^ 0x80000000u) : ~u);
}

// ---------------------------------------------------------------------------
// LDS loaders (blockDim.x == 256)
// ---------------------------------------------------------------------------
// Load 64x128 fp32 tile of A, convert to bf16, into LDS [64][SA]
__device__ __forceinline__ void load_a_lds(const float* __restrict__ A, int M, int mbase,
                                           unsigned short* As, int tid) {
#pragma unroll
    for (int i = 0; i < 8; ++i) {
        int chunk = tid + 256 * i;          // 2048 float4 chunks
        int o = chunk * 4;
        int row = o >> 7, col = o & 127;
        float4 v = make_float4(0.f, 0.f, 0.f, 0.f);
        int gr = mbase + row;
        if (gr < M) v = *(const float4*)(A + (size_t)gr * HCH + col);
        uint2 p;
        p.x = (unsigned)f2bf(v.x) | ((unsigned)f2bf(v.y) << 16);
        p.y = (unsigned)f2bf(v.z) | ((unsigned)f2bf(v.w) << 16);
        *(uint2*)(As + row * SA + col) = p;
    }
}
// Fallback: load 128x128 bf16 weight ([n][k] in global) into LDS [128][SA]
__device__ __forceinline__ void load_w_lds(const unsigned short* __restrict__ Wt,
                                           unsigned short* Ws, int tid) {
#pragma unroll
    for (int i = 0; i < 8; ++i) {
        int chunk = tid + 256 * i;          // 2048 uint4 chunks (8 bf16 each)
        int o = chunk * 8;
        int row = o >> 7, col = o & 127;
        *(uint4*)(Ws + row * SA + col) = *(const uint4*)(Wt + o);
    }
}

#if USE_TDM
typedef unsigned int tdm_u32x4 __attribute__((ext_vector_type(4)));
typedef int          tdm_i32x4 __attribute__((ext_vector_type(4)));
typedef int          tdm_i32x8 __attribute__((ext_vector_type(8)));

// DMA one 128x128 bf16 weight matrix (contiguous 32KB in global) into LDS,
// letting the TDM insert 16B of row padding every 128 bf16 to produce the
// [128][SA=136] layout. Issued by a single wave; waits on TENSORcnt.
// D# layout per CDNA5 ISA.
__device__ __forceinline__ void tdm_load_w(const unsigned short* __restrict__ Wt,
                                           unsigned short* Ws) {
    unsigned long long ga = (unsigned long long)(const void*)Wt;
    unsigned ldsa = (unsigned)(unsigned long long)(void*)Ws;   // LDS byte offset (addr[31:0])

    // group0: count=1 | lds_addr | global_addr[56:0] | type=2 (bits 127:126)
    tdm_u32x4 g0 = { 1u, ldsa, (unsigned)ga,
                     ((unsigned)(ga >> 32) & 0x01FFFFFFu) | 0x80000000u };

    // group1: data_size=1 (2B), pad_enable=1, pad_interval=5 (64 DWORDs = one
    // 128-bf16 row), pad_amount=3 (4 DWORDs = 8 bf16); tensor_dim0 = tile_dim0
    // = 16384 elems; tensor_dim1 = 1; dim0 stride = 16384.
    tdm_i32x8 g1 = { (int)((1u << 16) | (1u << 20) | (5u << 22) | (3u << 25)),
                     (int)(16384u << 16),   // tensor_dim0[15:0] at bits 63:48
                     (int)(1u << 16),       // tensor_dim0 hi=0, tensor_dim1 lo=1
                     (int)(16384u << 16),   // tensor_dim1 hi=0, tile_dim0=16384
                     0,                     // tile_dim1=0 (unused), tile_dim2=0
                     16384,                 // tensor_dim0_stride lo
                     0, 0 };

    tdm_i32x4 z4 = { 0, 0, 0, 0 };
#if __clang_major__ >= 23
    tdm_i32x8 z8 = { 0, 0, 0, 0, 0, 0, 0, 0 };
    __builtin_amdgcn_tensor_load_to_lds(g0, g1, z4, z4, z8, 0);
#else
    __builtin_amdgcn_tensor_load_to_lds(g0, g1, z4, z4, 0);
#endif
    __builtin_amdgcn_s_wait_tensorcnt(0);
}
#endif  // USE_TDM

// Weight load dispatch: TDM (single wave) when available, else cooperative copy.
__device__ __forceinline__ void weight_to_lds(const unsigned short* __restrict__ Wt,
                                              unsigned short* Ws, int tid) {
#if USE_TDM
    if (tid < 32) tdm_load_w(Wt, Ws);
#else
    load_w_lds(Wt, Ws, tid);
#endif
}

// ---------------------------------------------------------------------------
// One wave computes a 16x64 output tile: rows [mrow0,mrow0+16), col tiles nt0..nt0+3
// K = 128 (4 WMMA k-steps). A in LDS [64][SA] bf16 row-major, W in LDS [128][SA]
// stored as [n][k] so B fragments are contiguous in k.
// ---------------------------------------------------------------------------
__device__ __forceinline__ void wave_mma_64(const unsigned short* As, const unsigned short* Ws,
                                            int mrow0, int nt0, int l16, int half, v8f* acc) {
#pragma unroll
    for (int kk = 0; kk < 128; kk += 32) {
        BFrag a;
        // A 16x32 bf16 layout: lanes 0-15 hold K kk+0..7 & kk+16..23; lanes 16-31 +8
        const unsigned short* ap = As + (mrow0 + l16) * SA + kk + half * 8;
        a.u[0] = *(const uint4*)(ap);
        a.u[1] = *(const uint4*)(ap + 16);
#pragma unroll
        for (int t = 0; t < 4; ++t) {
            BFrag b;
            // B 32x16 layout: col n = lane%16, lanes 0-15 K kk+0..7 & kk+16..23; lanes 16-31 +8
            const unsigned short* bp = Ws + ((nt0 + t) * 16 + l16) * SA + kk + half * 8;
            b.u[0] = *(const uint4*)(bp);
            b.u[1] = *(const uint4*)(bp + 16);
            acc[t] = __builtin_amdgcn_wmma_f32_16x16x32_bf16(
                false, a.v, false, b.v, (short)0, acc[t], false, false);
        }
    }
}

// ---------------------------------------------------------------------------
// Generic C[M,128] = act(A[M,128] @ W[128,128] + bias) via WMMA
// ---------------------------------------------------------------------------
__global__ __launch_bounds__(256) void gemm128_kernel(const float* __restrict__ A,
                                                      const unsigned short* __restrict__ Wt,
                                                      const float* __restrict__ bias,
                                                      float* __restrict__ C, int M, int act) {
    __shared__ unsigned short As[64 * SA];
    __shared__ unsigned short Ws[128 * SA];
    const int tid = threadIdx.x;
    const int mbase = blockIdx.x * 64;

    weight_to_lds(Wt, Ws, tid);
    load_a_lds(A, M, mbase, As, tid);
    __syncthreads();

    const int wv = tid >> 5, lane = tid & 31, l16 = lane & 15, half = lane >> 4;
    const int mrow0 = (wv & 3) * 16, nt0 = (wv >> 2) * 4;

    v8f acc[4];
#pragma unroll
    for (int t = 0; t < 4; ++t)
#pragma unroll
        for (int r = 0; r < 8; ++r) acc[t][r] = 0.f;

    wave_mma_64(As, Ws, mrow0, nt0, l16, half, acc);

#pragma unroll
    for (int t = 0; t < 4; ++t) {
        int n = (nt0 + t) * 16 + l16;
        float bv = bias ? bias[n] : 0.f;
#pragma unroll
        for (int r = 0; r < 8; ++r) {
            int row = mbase + mrow0 + r + 8 * half;   // C/D: lanes16-31 -> M = r+8
            if (row < M) {
                float v = acc[t][r] + bv;
                if (act) v = leaky(v);
                C[(size_t)row * HCH + n] = v;
            }
        }
    }
}

// ---------------------------------------------------------------------------
// Fused edge MLP + message passing:
//   e  = leaky(leaky(bonds@W1+b1)@W2+b2)          (two WMMA GEMMs, tile in LDS)
//   inner[seg[e]] += nodeproj[idx[e]] * e          (atomic scatter)
// The intermediate activation never touches HBM.
// ---------------------------------------------------------------------------
__global__ __launch_bounds__(256) void edge_mlp_message_kernel(
    const float* __restrict__ bonds, const unsigned short* __restrict__ W1t,
    const unsigned short* __restrict__ W2t, const float* __restrict__ b1,
    const float* __restrict__ b2, const float* __restrict__ nodeproj,
    const int* __restrict__ seg, const int* __restrict__ idx,
    float* __restrict__ inner_out, int E) {
    __shared__ unsigned short As[64 * SA];    // bonds tile, then layer1 act, then layer2 act
    __shared__ unsigned short Ws[128 * SA];   // W1, then W2
    __shared__ int sIdx[64], sSeg[64];
    __shared__ float sB1[128], sB2[128];

    const int tid = threadIdx.x;
    const int ebase = blockIdx.x * 64;

    weight_to_lds(W1t, Ws, tid);
    load_a_lds(bonds, E, ebase, As, tid);
    if (tid < 128) { sB1[tid] = b1[tid]; sB2[tid] = b2[tid]; }
    if (tid < 64) {
        int ge = ebase + tid;
        sIdx[tid] = (ge < E) ? idx[ge] : 0;
        sSeg[tid] = (ge < E) ? seg[ge] : 0;
    }
    __syncthreads();

    const int wv = tid >> 5, lane = tid & 31, l16 = lane & 15, half = lane >> 4;
    const int mrow0 = (wv & 3) * 16, nt0 = (wv >> 2) * 4;

    v8f acc[4];
#pragma unroll
    for (int t = 0; t < 4; ++t)
#pragma unroll
        for (int r = 0; r < 8; ++r) acc[t][r] = 0.f;
    wave_mma_64(As, Ws, mrow0, nt0, l16, half, acc);
    __syncthreads();                                   // everyone done reading As / Ws

    // write layer-1 activation back into As (WAR protected by barrier above)
#pragma unroll
    for (int t = 0; t < 4; ++t) {
        int n = (nt0 + t) * 16 + l16;
#pragma unroll
        for (int r = 0; r < 8; ++r) {
            int lr = mrow0 + r + 8 * half;
            As[lr * SA + n] = f2bf(leaky(acc[t][r] + sB1[n]));
        }
    }
    weight_to_lds(W2t, Ws, tid);                       // swap in W2 (TDM when available)
    __syncthreads();

#pragma unroll
    for (int t = 0; t < 4; ++t)
#pragma unroll
        for (int r = 0; r < 8; ++r) acc[t][r] = 0.f;
    wave_mma_64(As, Ws, mrow0, nt0, l16, half, acc);
    __syncthreads();

#pragma unroll
    for (int t = 0; t < 4; ++t) {
        int n = (nt0 + t) * 16 + l16;
#pragma unroll
        for (int r = 0; r < 8; ++r) {
            int lr = mrow0 + r + 8 * half;
            As[lr * SA + n] = f2bf(leaky(acc[t][r] + sB2[n]));
        }
    }
    __syncthreads();

    // scatter messages: 64 edges x 128 channels, 32 per thread
#pragma unroll
    for (int i = 0; i < 32; ++i) {
        int elem = tid + 256 * i;
        int le = elem >> 7, c = elem & 127;
        int ge = ebase + le;
        if (ge < E) {
            float msg = nodeproj[(size_t)sIdx[le] * HCH + c] * bf2f(As[le * SA + c]);
            atomicAdd(inner_out + (size_t)sSeg[le] * HCH + c, msg);
        }
    }
}

// ---------------------------------------------------------------------------
// Small helper kernels
// ---------------------------------------------------------------------------
__global__ void zero_f_kernel(float* p, long n) {
    long i = (long)blockIdx.x * blockDim.x + threadIdx.x;
    long stride = (long)gridDim.x * blockDim.x;
    for (; i < n; i += stride) p[i] = 0.f;
}

// fp32 [K=128][N=128] -> bf16 [N][K] (transposed for contiguous B fragments)
__global__ void cvt_w_kernel(const float* __restrict__ src, unsigned short* __restrict__ dst) {
    int t = blockIdx.x * 256 + threadIdx.x;
    if (t < 128 * 128) {
        int k = t >> 7, n = t & 127;
        dst[n * 128 + k] = f2bf(src[t]);
    }
}

// t[e] = dot(KL[il[e]], KR[ir[e]]) over 128 channels; one wave per edge
__global__ void edge_dot_kernel(const float* __restrict__ KL, const float* __restrict__ KR,
                                const int* __restrict__ il, const int* __restrict__ ir,
                                float* __restrict__ T, int E) {
    int gtid = blockIdx.x * 256 + threadIdx.x;
    int e = gtid >> 5, lane = gtid & 31;
    if (e >= E) return;
    int a = il[e], b = ir[e];
    float4 u = *(const float4*)(KL + (size_t)a * HCH + lane * 4);
    float4 v = *(const float4*)(KR + (size_t)b * HCH + lane * 4);
    float d = u.x * v.x + u.y * v.y + u.z * v.z + u.w * v.w;
#pragma unroll
    for (int off = 16; off; off >>= 1) d += __shfl_xor(d, off, 32);
    if (lane == 0) T[e] = d;
}

__global__ void seg_max_kernel(const float* __restrict__ T, const int* __restrict__ il,
                               const int* __restrict__ ir, unsigned* ZML, unsigned* ZMR, int E) {
    int e = blockIdx.x * 256 + threadIdx.x;
    if (e >= E) return;
    unsigned enc = fflip(T[e]);
    atomicMax(ZML + il[e], enc);
    atomicMax(ZMR + ir[e], enc);
}

__global__ void exp_sum_kernel(const float* __restrict__ T, const int* __restrict__ il,
                               const int* __restrict__ ir, const unsigned* __restrict__ ZML,
                               const unsigned* __restrict__ ZMR, float* __restrict__ EEL,
                               float* __restrict__ EER, float* ZSL, float* ZSR, int E) {
    int e = blockIdx.x * 256 + threadIdx.x;
    if (e >= E) return;
    float t = T[e];
    float ml = funflip(ZML[il[e]]);
    float mr = funflip(ZMR[ir[e]]);
    float el = __expf((t - ml) * INV_TEMP);
    float er = __expf((t - mr) * INV_TEMP);
    EEL[e] = el;
    EER[e] = er;
    atomicAdd(ZSL + il[e], el);
    atomicAdd(ZSR + ir[e], er);
}

// m_l[il] += a_l * vr[ir];  m_r[ir] += a_r * vl[il]   (one thread per edge-channel)
__global__ void attn_scatter_kernel(const float* __restrict__ EEL, const float* __restrict__ EER,
                                    const float* __restrict__ ZSL, const float* __restrict__ ZSR,
                                    const float* __restrict__ VLp, const float* __restrict__ VRp,
                                    const int* __restrict__ il, const int* __restrict__ ir,
                                    float* ML, float* MR, int E) {
    long gtid = (long)blockIdx.x * 256 + threadIdx.x;
    int e = (int)(gtid >> 7), c = (int)(gtid & 127);
    if (e >= E) return;
    int a = il[e], b = ir[e];
    float al = EEL[e] / (ZSL[a] + 1e-16f);
    float ar = EER[e] / (ZSR[b] + 1e-16f);
    atomicAdd(ML + (size_t)a * HCH + c, al * VRp[(size_t)b * HCH + c]);
    atomicAdd(MR + (size_t)b * HCH + c, ar * VLp[(size_t)a * HCH + c]);
}

// x = atom + inner + outer; LayerNorm -> out. One wave per row.
__global__ __launch_bounds__(256) void layernorm_kernel(
    const float* __restrict__ atom, const float* __restrict__ inner,
    const float* __restrict__ outer, const float* __restrict__ gamma,
    const float* __restrict__ beta, float* __restrict__ out, int Nrows) {
    int wv = threadIdx.x >> 5, lane = threadIdx.x & 31;
    int row = blockIdx.x * 8 + wv;
    if (row >= Nrows) return;
    size_t base = (size_t)row * HCH;
    int c0 = lane * 4;
    float x[4], s = 0.f, s2 = 0.f;
#pragma unroll
    for (int i = 0; i < 4; ++i) {
        float v = atom[base + c0 + i] + inner[base + c0 + i] + outer[base + c0 + i];
        x[i] = v;
        s += v;
        s2 += v * v;
    }
#pragma unroll
    for (int off = 16; off; off >>= 1) {
        s += __shfl_xor(s, off, 32);
        s2 += __shfl_xor(s2, off, 32);
    }
    float mu = s * (1.f / 128.f);
    float var = s2 * (1.f / 128.f) - mu * mu;
    float rs = rsqrtf(var + LN_EPS);
#pragma unroll
    for (int i = 0; i < 4; ++i)
        out[base + c0 + i] = (x[i] - mu) * rs * gamma[c0 + i] + beta[c0 + i];
}

// graph[seg_mol[i]] += a[i]   (one thread per atom-channel)
__global__ void mol_scatter_kernel(const float* __restrict__ A, const int* __restrict__ segmol,
                                   float* out, int Nrows) {
    long gtid = (long)blockIdx.x * 256 + threadIdx.x;
    int i = (int)(gtid >> 7), c = (int)(gtid & 127);
    if (i >= Nrows) return;
    atomicAdd(out + (size_t)segmol[i] * HCH + c, A[(size_t)i * HCH + c]);
}

// ---------------------------------------------------------------------------
// Host-side orchestration
// ---------------------------------------------------------------------------
extern "C" void kernel_launch(void* const* d_in, const int* in_sizes, int n_in,
                              void* d_out, int out_size, void* d_ws, size_t ws_size,
                              hipStream_t stream) {
    (void)n_in; (void)ws_size;

    const int* segMolL = (const int*)d_in[0];
    const float* atomL = (const float*)d_in[1];
    const float* bondL = (const float*)d_in[2];
    const int* innSegL = (const int*)d_in[3];
    const int* innIdxL = (const int*)d_in[4];
    const int* outSegL = (const int*)d_in[5];
    const int* segMolR = (const int*)d_in[7];
    const float* atomR = (const float*)d_in[8];
    const float* bondR = (const float*)d_in[9];
    const int* innSegR = (const int*)d_in[10];
    const int* innIdxR = (const int*)d_in[11];
    const int* outSegR = (const int*)d_in[12];
    const float* W_node = (const float*)d_in[14];
    const float* W_e1 = (const float*)d_in[15];
    const float* b_e1 = (const float*)d_in[16];
    const float* W_e2 = (const float*)d_in[17];
    const float* b_e2 = (const float*)d_in[18];
    const float* W_key = (const float*)d_in[19];
    const float* W_val = (const float*)d_in[20];
    const float* W_att = (const float*)d_in[21];
    const float* b_att = (const float*)d_in[22];
    const float* gam = (const float*)d_in[23];
    const float* bet = (const float*)d_in[24];
    const float* W_read = (const float*)d_in[25];
    const float* b_read = (const float*)d_in[26];

    const int N = in_sizes[1] / HCH;        // 20000
    const int EIN = in_sizes[2] / HCH;      // 262144
    const int EOUT = in_sizes[5];           // 262144
    const int NMOL = out_size / (2 * HCH);  // 512
    const long NH = (long)N * HCH;

    // ---- workspace layout ----
    unsigned short* wt = (unsigned short*)d_ws;     // 7 bf16 weight mats, [n][k]
    unsigned short* wtN = wt + 0 * 16384;
    unsigned short* wtE1 = wt + 1 * 16384;
    unsigned short* wtE2 = wt + 2 * 16384;
    unsigned short* wtK = wt + 3 * 16384;
    unsigned short* wtV = wt + 4 * 16384;
    unsigned short* wtA = wt + 5 * 16384;
    unsigned short* wtR = wt + 6 * 16384;
    float* F = (float*)(wt + 7 * 16384);
    float* KL = F + 0 * NH;
    float* KR = F + 1 * NH;
    float* VL = F + 2 * NH;
    float* VR = F + 3 * NH;
    float* NLp = F + 4 * NH;
    float* NRp = F + 5 * NH;
    float* INL = F + 6 * NH;
    float* INR = F + 7 * NH;
    float* XLNL = F + 8 * NH;
    float* XLNR = F + 9 * NH;
    float* ARL = F + 10 * NH;
    float* ARR = F + 11 * NH;
    float* T = F + 12 * NH;
    float* EEL = T + EOUT;
    float* EER = T + 2 * EOUT;
    unsigned* ZML = (unsigned*)(T + 3 * (long)EOUT);
    unsigned* ZMR = ZML + N;
    float* ZSL = (float*)(ZMR + N);
    float* ZSR = ZSL + N;
    // buffer reuse (sequenced on stream): attention accumulators over K bufs,
    // attention output over V bufs (V consumed by attn_scatter first).
    float* MACCL = KL;
    float* MACCR = KR;
    float* OUTL = VL;
    float* OUTR = VR;

    float* outG = (float*)d_out;
    float* outL = outG;
    float* outR = outG + (long)NMOL * HCH;

    const int gN64 = (N + 63) / 64;
    const int gE64 = (EIN + 63) / 64;

    // 1) zero accumulators / stats / output
    zero_f_kernel<<<2048, 256, 0, stream>>>(INL, 2 * NH);
    zero_f_kernel<<<64, 256, 0, stream>>>((float*)ZML, 4L * N);
    zero_f_kernel<<<256, 256, 0, stream>>>(outG, 2L * NMOL * HCH);

    // 2) convert + transpose weights fp32 -> bf16 [n][k]
    cvt_w_kernel<<<64, 256, 0, stream>>>(W_node, wtN);
    cvt_w_kernel<<<64, 256, 0, stream>>>(W_e1, wtE1);
    cvt_w_kernel<<<64, 256, 0, stream>>>(W_e2, wtE2);
    cvt_w_kernel<<<64, 256, 0, stream>>>(W_key, wtK);
    cvt_w_kernel<<<64, 256, 0, stream>>>(W_val, wtV);
    cvt_w_kernel<<<64, 256, 0, stream>>>(W_att, wtA);
    cvt_w_kernel<<<64, 256, 0, stream>>>(W_read, wtR);

    // 3) dense projections (WMMA): K/V/node for both sides
    gemm128_kernel<<<gN64, 256, 0, stream>>>(atomL, wtK, nullptr, KL, N, 0);
    gemm128_kernel<<<gN64, 256, 0, stream>>>(atomR, wtK, nullptr, KR, N, 0);
    gemm128_kernel<<<gN64, 256, 0, stream>>>(atomL, wtV, nullptr, VL, N, 0);
    gemm128_kernel<<<gN64, 256, 0, stream>>>(atomR, wtV, nullptr, VR, N, 0);
    gemm128_kernel<<<gN64, 256, 0, stream>>>(atomL, wtN, nullptr, NLp, N, 0);
    gemm128_kernel<<<gN64, 256, 0, stream>>>(atomR, wtN, nullptr, NRp, N, 0);

    // 4) co-attention logits (one wave per pair)
    edge_dot_kernel<<<(EOUT * 32 + 255) / 256, 256, 0, stream>>>(KL, KR, outSegL, outSegR, T, EOUT);

    // 5) K bufs now free -> become attention accumulators
    zero_f_kernel<<<2048, 256, 0, stream>>>(MACCL, 2 * NH);

    // 6-7) ragged softmax (max via encoded atomicMax, sum via atomicAdd)
    seg_max_kernel<<<(EOUT + 255) / 256, 256, 0, stream>>>(T, outSegL, outSegR, ZML, ZMR, EOUT);
    exp_sum_kernel<<<(EOUT + 255) / 256, 256, 0, stream>>>(T, outSegL, outSegR, ZML, ZMR,
                                                           EEL, EER, ZSL, ZSR, EOUT);

    // 8) weighted value scatter (both directions)
    attn_scatter_kernel<<<(int)(((long)EOUT * HCH + 255) / 256), 256, 0, stream>>>(
        EEL, EER, ZSL, ZSR, VL, VR, outSegL, outSegR, MACCL, MACCR, EOUT);

    // 9) fused edge MLP (2x WMMA GEMM in LDS) + message scatter
    edge_mlp_message_kernel<<<gE64, 256, 0, stream>>>(bondL, wtE1, wtE2, b_e1, b_e2, NLp,
                                                      innSegL, innIdxL, INL, EIN);
    edge_mlp_message_kernel<<<gE64, 256, 0, stream>>>(bondR, wtE1, wtE2, b_e1, b_e2, NRp,
                                                      innSegR, innIdxR, INR, EIN);

    // 10) attention output projection (V bufs are free now)
    gemm128_kernel<<<gN64, 256, 0, stream>>>(MACCL, wtA, b_att, OUTL, N, 1);
    gemm128_kernel<<<gN64, 256, 0, stream>>>(MACCR, wtA, b_att, OUTR, N, 1);

    // 11) residual add + LayerNorm
    layernorm_kernel<<<(N + 7) / 8, 256, 0, stream>>>(atomL, INL, OUTL, gam, bet, XLNL, N);
    layernorm_kernel<<<(N + 7) / 8, 256, 0, stream>>>(atomR, INR, OUTR, gam, bet, XLNR, N);

    // 12) readout projection
    gemm128_kernel<<<gN64, 256, 0, stream>>>(XLNL, wtR, b_read, ARL, N, 1);
    gemm128_kernel<<<gN64, 256, 0, stream>>>(XLNR, wtR, b_read, ARR, N, 1);

    // 13) per-molecule segment sum into output
    mol_scatter_kernel<<<(int)((NH + 255) / 256), 256, 0, stream>>>(ARL, segMolL, outL, N);
    mol_scatter_kernel<<<(int)((NH + 255) / 256), 256, 0, stream>>>(ARR, segMolR, outR, N);
}
